// SwinTransformer_712964571505
// MI455X (gfx1250) — compile-verified, matrix operand order checked by reading, not run on hardware
//
#include <hip/hip_runtime.h>
#include <hip/hip_bf16.h>

typedef _Float16 h16;
typedef __attribute__((ext_vector_type(16))) _Float16 v16h;
typedef __attribute__((ext_vector_type(8)))  float    v8f;
typedef __attribute__((ext_vector_type(4)))  int      v4i;

#if __has_builtin(__builtin_amdgcn_global_load_async_to_lds_b128)
#define ASYNC_LDS 1
typedef __attribute__((address_space(1))) v4i* gv4i_p;   // global (__device__)
typedef __attribute__((address_space(3))) v4i* lv4i_p;   // LDS (__shared__)
#else
#define ASYNC_LDS 0
#endif

__device__ __forceinline__ void wait_async_all() {
#if ASYNC_LDS
#if __has_builtin(__builtin_amdgcn_s_wait_asynccnt)
  __builtin_amdgcn_s_wait_asynccnt(0);
#else
  asm volatile("s_wait_asynccnt 0x0" ::: "memory");
#endif
#endif
}

// ---------------------------------------------------------------------------
// WMMA helpers (CDNA5 wave32).  A operand: 16x32 f16, one row per lane
// (lanes 0-15 rows, lane groups hold K-halves per ISA layout).  B operand is
// loaded through the same "row=N, k" view (i.e. Bt[n][k]).
// D/C layout: lane = column (lane&15), VGPR v = row v + 8*(lane>>4).
// ---------------------------------------------------------------------------
__device__ __forceinline__ v8f wmma_f16(v16h a, v16h b, v8f c) {
  return __builtin_amdgcn_wmma_f32_16x16x32_f16(false, a, false, b, (short)0, c,
                                                false, false);
}

// Generic strided 16x32 tile loader matching the documented 16-bit A layout:
// lane<16: elems 0..7 -> K k0+0..7, elems 8..15 -> K k0+16..23
// lane>=16: elems 0..7 -> K k0+8..15, elems 8..15 -> K k0+24..31
__device__ __forceinline__ v16h tile_ld(const h16* __restrict__ base, int rs, int ks,
                                        int row0, int k0, int lane) {
  const int r  = row0 + (lane & 15);
  const int kb = k0 + ((lane >> 4) << 3);
  const h16* p = base + (size_t)r * rs;
  v16h a;
#pragma unroll
  for (int e = 0; e < 8; ++e) a[e] = p[(size_t)(kb + e) * ks];
#pragma unroll
  for (int e = 0; e < 8; ++e) a[8 + e] = p[(size_t)(kb + 16 + e) * ks];
  return a;
}

__device__ __forceinline__ float gelu_exact(float v) {
  return 0.5f * v * (1.0f + erff(v * 0.70710678118654752f));
}

// ---------------------------------------------------------------------------
// Patch embed: 4x4x3 -> 96 matmul + bias + LN.  One block (128 thr) per token.
// ---------------------------------------------------------------------------
__global__ __launch_bounds__(128)
void k_patch_embed(const float* __restrict__ x, const float* __restrict__ w,
                   const float* __restrict__ bias, const float* __restrict__ g,
                   const float* __restrict__ be, float* __restrict__ X) {
  const int t  = blockIdx.x;                 // 8*56*56 tokens
  const int bb = t / (56 * 56);
  const int rm = t % (56 * 56);
  const int hp = rm / 56, wp = rm % 56;
  const int tid = threadIdx.x;
  __shared__ float patch[48];
  __shared__ float val[96];
  __shared__ float s1[128], s2[128];
  if (tid < 48) {
    int i = tid / 12, j = (tid / 3) % 4, ch = tid % 3;
    patch[tid] = x[(((size_t)bb * 3 + ch) * 224 + hp * 4 + i) * 224 + wp * 4 + j];
  }
  __syncthreads();
  float v = 0.f;
  if (tid < 96) {
    v = bias[tid];
#pragma unroll 4
    for (int f = 0; f < 48; ++f) v += patch[f] * w[f * 96 + tid];
    val[tid] = v;
  }
  s1[tid] = (tid < 96) ? v : 0.f;
  s2[tid] = (tid < 96) ? v * v : 0.f;
  __syncthreads();
  for (int st = 64; st > 0; st >>= 1) {
    if (tid < st) { s1[tid] += s1[tid + st]; s2[tid] += s2[tid + st]; }
    __syncthreads();
  }
  float mean = s1[0] / 96.f;
  float var  = s2[0] / 96.f - mean * mean;
  float inv  = rsqrtf(var + 1e-5f);
  if (tid < 96)
    X[(size_t)t * 96 + tid] = (val[tid] - mean) * inv * g[tid] + be[tid];
}

// ---------------------------------------------------------------------------
// LN + (shifted) window partition into padded [Bn, 64, C] f16 buffer.
// ---------------------------------------------------------------------------
__global__ __launch_bounds__(128)
void k_ln_part(const float* __restrict__ X, const float* __restrict__ g,
               const float* __restrict__ b, h16* __restrict__ XW,
               int Hs, int C, int shift, int nWs) {
  const int id = blockIdx.x;
  const int win = id >> 6, p = id & 63;
  const int tid = threadIdx.x;
  h16* dst = XW + ((size_t)win * 64 + p) * C;
  if (p >= 49) { for (int c = tid; c < C; c += 128) dst[c] = (h16)0.f; return; }
  const int nW2 = nWs * nWs;
  const int bb = win / nW2, wij = win % nW2;
  const int wi = wij / nWs, wj = wij % nWs;
  const int r = p / 7, cc = p % 7;
  const int hh = (wi * 7 + r + shift) % Hs;
  const int ww = (wj * 7 + cc + shift) % Hs;
  const float* xp = X + (((size_t)bb * Hs + hh) * Hs + ww) * C;
  __shared__ float s1[128], s2[128];
  float a = 0.f, q = 0.f;
  for (int c = tid; c < C; c += 128) { float v = xp[c]; a += v; q += v * v; }
  s1[tid] = a; s2[tid] = q; __syncthreads();
  for (int st = 64; st > 0; st >>= 1) {
    if (tid < st) { s1[tid] += s1[tid + st]; s2[tid] += s2[tid + st]; }
    __syncthreads();
  }
  float mean = s1[0] / C, var = s2[0] / C - mean * mean;
  float inv = rsqrtf(var + 1e-5f);
  for (int c = tid; c < C; c += 128)
    dst[c] = (h16)((xp[c] - mean) * inv * g[c] + b[c]);
}

// LN over tokens (no windowing), output f16 [T, C]
__global__ __launch_bounds__(128)
void k_ln_tok(const float* __restrict__ X, const float* __restrict__ g,
              const float* __restrict__ b, h16* __restrict__ out, int C) {
  const int t = blockIdx.x;
  const int tid = threadIdx.x;
  const float* xp = X + (size_t)t * C;
  h16* dst = out + (size_t)t * C;
  __shared__ float s1[128], s2[128];
  float a = 0.f, q = 0.f;
  for (int c = tid; c < C; c += 128) { float v = xp[c]; a += v; q += v * v; }
  s1[tid] = a; s2[tid] = q; __syncthreads();
  for (int st = 64; st > 0; st >>= 1) {
    if (tid < st) { s1[tid] += s1[tid + st]; s2[tid] += s2[tid + st]; }
    __syncthreads();
  }
  float mean = s1[0] / C, var = s2[0] / C - mean * mean;
  float inv = rsqrtf(var + 1e-5f);
  for (int c = tid; c < C; c += 128)
    dst[c] = (h16)((xp[c] - mean) * inv * g[c] + b[c]);
}

// ---------------------------------------------------------------------------
// Tiled f16 WMMA GEMM:  Y[M,N] = op(X[M,K] * W[K,N] + bias)
// X f16 in global (A staged to LDS via gfx1250 async-to-LDS when available),
// W f32 converted to f16 while staging.  flags: bit0 GELU, bit1 f32 output.
// Block 256 thr = 8 waves (4x2), tile 128x64, K step 32.
// ---------------------------------------------------------------------------
#define BM 128
#define BN 64
#define BK 32
__global__ __launch_bounds__(256)
void k_gemm(const h16* __restrict__ Xin, const float* __restrict__ W,
            const float* __restrict__ bias, h16* __restrict__ Yh,
            float* __restrict__ Yf, int M, int N, int K, int flags) {
  __shared__ __align__(16) h16 As[BM][BK];
  __shared__ __align__(16) h16 Bs[BN][BK];
  const int tid = threadIdx.x;
  const int lane = tid & 31;
  const int wave = tid >> 5;
  const int m0 = blockIdx.x * BM;
  const int n0 = blockIdx.y * BN;
  const int wm = (wave >> 1) * 32;
  const int wn = (wave & 1) * 32;
  v8f acc[2][2] = {};
  for (int kk = 0; kk < K; kk += BK) {
    // ---- stage A: 512 chunks of 8 halves (16B), 2 per thread -------------
#pragma unroll
    for (int qq = 0; qq < 2; ++qq) {
      int q = tid + qq * 256;
      int r = q >> 2, c = (q & 3) * 8;
#if ASYNC_LDS
      if (m0 + r < M) {
        __builtin_amdgcn_global_load_async_to_lds_b128(
            (gv4i_p)(Xin + (size_t)(m0 + r) * K + kk + c),
            (lv4i_p)(&As[r][c]), 0, 0);
      } else {
        *reinterpret_cast<uint4*>(&As[r][c]) = make_uint4(0u, 0u, 0u, 0u);
      }
#else
      uint4 v = make_uint4(0u, 0u, 0u, 0u);
      if (m0 + r < M)
        v = *reinterpret_cast<const uint4*>(Xin + (size_t)(m0 + r) * K + kk + c);
      *reinterpret_cast<uint4*>(&As[r][c]) = v;
#endif
    }
    // ---- stage B (transpose to [n][k], f32 -> f16, packed b32 stores) ----
    {
      int n = tid & 63, j0 = (tid >> 6) * 8;
      bool ok = (n0 + n) < N;
#pragma unroll
      for (int j = j0; j < j0 + 8; j += 2) {
        float v0 = ok ? W[(size_t)(kk + j) * N + n0 + n] : 0.f;
        float v1 = ok ? W[(size_t)(kk + j + 1) * N + n0 + n] : 0.f;
        union { h16 h[2]; unsigned u; } pk;
        pk.h[0] = (h16)v0; pk.h[1] = (h16)v1;
        *reinterpret_cast<unsigned*>(&Bs[n][j]) = pk.u;
      }
      if (ok && kk + BK < K)   // prefetch next K-slice of W (global_prefetch)
        __builtin_prefetch(&W[(size_t)(kk + BK + j0) * N + n0 + n], 0, 1);
    }
    wait_async_all();
    __syncthreads();
    v16h a0 = tile_ld(&As[0][0], BK, 1, wm, 0, lane);
    v16h a1 = tile_ld(&As[0][0], BK, 1, wm + 16, 0, lane);
    v16h b0 = tile_ld(&Bs[0][0], BK, 1, wn, 0, lane);
    v16h b1 = tile_ld(&Bs[0][0], BK, 1, wn + 16, 0, lane);
    acc[0][0] = wmma_f16(a0, b0, acc[0][0]);
    acc[0][1] = wmma_f16(a0, b1, acc[0][1]);
    acc[1][0] = wmma_f16(a1, b0, acc[1][0]);
    acc[1][1] = wmma_f16(a1, b1, acc[1][1]);
    __syncthreads();
  }
  const int hi = lane >> 4, ln = lane & 15;
#pragma unroll
  for (int i = 0; i < 2; ++i)
#pragma unroll
    for (int j = 0; j < 2; ++j) {
      int cn = n0 + wn + j * 16 + ln;
      if (cn >= N) continue;
      float bval = bias ? bias[cn] : 0.f;
#pragma unroll
      for (int v = 0; v < 8; ++v) {
        int rm = m0 + wm + i * 16 + hi * 8 + v;
        if (rm >= M) continue;
        float val = acc[i][j][v] + bval;
        if (flags & 1) val = gelu_exact(val);
        if (flags & 2) Yf[(size_t)rm * N + cn] = val;
        else           Yh[(size_t)rm * N + cn] = (h16)val;
      }
    }
}

// ---------------------------------------------------------------------------
// Windowed attention: one wave per (window, head); head_dim = 32, N = 49
// padded to 64.  S = Q*Kt (16 WMMAs), softmax (+rel bias +shift mask) in the
// accumulator layout, P staged through LDS, O = P*V (16 WMMAs).
// ---------------------------------------------------------------------------
__device__ __forceinline__ int reg3(int h, int Hs) {
  return (h < Hs - 7) ? 0 : (h < Hs - 3 ? 1 : 2);
}

__global__ void k_attn(const h16* __restrict__ qkv, h16* __restrict__ aout,
                       const float* __restrict__ rel, int C, int nH, int shift,
                       int nWs) {
  extern __shared__ h16 Pls[];                 // nH * 64 * 64 f16
  const int win = blockIdx.x;
  const int head = threadIdx.x >> 5;
  const int lane = threadIdx.x & 31;
  h16* P = Pls + (size_t)head * 64 * 64;
  const int C3 = 3 * C;
  const h16* qb = qkv + (size_t)win * 64 * C3 + head * 32;
  const h16* kb = qb + C;
  const h16* vb = qb + 2 * C;
  const int nW2 = nWs * nWs;
  const int wij = win % nW2;
  const int wi = wij / nWs, wj = wij % nWs;
  const int Hs = nWs * 7;
  const int hi = lane >> 4, ln = lane & 15;
  const float scale = 0.17677669529663689f;    // 1/sqrt(32)

  // ---- scores S[64][64]: K tiles hoisted, loaded once ------------------
  v16h kt[4];
#pragma unroll
  for (int ni = 0; ni < 4; ++ni) kt[ni] = tile_ld(kb, C3, 1, ni * 16, 0, lane);
  v8f s[4][4];
#pragma unroll
  for (int mi = 0; mi < 4; ++mi)
#pragma unroll
    for (int ni = 0; ni < 4; ++ni) s[mi][ni] = (v8f){0,0,0,0,0,0,0,0};
#pragma unroll
  for (int mi = 0; mi < 4; ++mi) {
    v16h qa = tile_ld(qb, C3, 1, mi * 16, 0, lane);
#pragma unroll
    for (int ni = 0; ni < 4; ++ni)
      s[mi][ni] = wmma_f16(qa, kt[ni], s[mi][ni]);
  }
  // ---- key-side per-lane coords, hoisted -------------------------------
  int qr[4], qc[4], rgq[4];
  bool kval[4];
#pragma unroll
  for (int ni = 0; ni < 4; ++ni) {
    int cq = ni * 16 + ln;
    kval[ni] = cq < 49;
    qr[ni] = cq / 7; qc[ni] = cq % 7;
    rgq[ni] = reg3(wi * 7 + qr[ni], Hs) * 3 + reg3(wj * 7 + qc[ni], Hs);
  }
  // ---- bias + mask + softmax (query coords hoisted per (mi,v)) ---------
#pragma unroll
  for (int mi = 0; mi < 4; ++mi)
#pragma unroll
    for (int v = 0; v < 8; ++v) {
      int rq = mi * 16 + hi * 8 + v;
      bool rv = rq < 49;
      int pr = rq / 7, pc = rq % 7;
      int rgp = reg3(wi * 7 + pr, Hs) * 3 + reg3(wj * 7 + pc, Hs);
      const float* relq = rel + head;
#pragma unroll
      for (int ni = 0; ni < 4; ++ni) {
        float val;
        if (rv && kval[ni]) {
          int ridx = (pr - qr[ni] + 6) * 13 + (pc - qc[ni] + 6);
          val = s[mi][ni][v] * scale + relq[ridx * nH];
          if (shift > 0 && rgp != rgq[ni]) val -= 100.f;
        } else {
          val = -1e30f;                         // exclude padded keys
        }
        s[mi][ni][v] = val;
      }
      // row softmax: reduce across ni (regs) and 16 lanes of this group
      float m = -1e30f;
#pragma unroll
      for (int ni = 0; ni < 4; ++ni) m = fmaxf(m, s[mi][ni][v]);
      for (int off = 8; off >= 1; off >>= 1) m = fmaxf(m, __shfl_xor(m, off));
      float sum = 0.f;
#pragma unroll
      for (int ni = 0; ni < 4; ++ni) {
        float e = __expf(s[mi][ni][v] - m);
        s[mi][ni][v] = e;
        sum += e;
      }
      for (int off = 8; off >= 1; off >>= 1) sum += __shfl_xor(sum, off);
      float inv = 1.f / sum;
#pragma unroll
      for (int ni = 0; ni < 4; ++ni) s[mi][ni][v] *= inv;
    }
  // ---- P -> LDS (re-layout for A operand) ------------------------------
#pragma unroll
  for (int mi = 0; mi < 4; ++mi)
#pragma unroll
    for (int ni = 0; ni < 4; ++ni)
#pragma unroll
      for (int v = 0; v < 8; ++v) {
        int rq = mi * 16 + hi * 8 + v, cq = ni * 16 + ln;
        P[rq * 64 + cq] = (h16)s[mi][ni][v];
      }
  // ---- O = P * V  (M=64, N=32, K=64) -----------------------------------
  v8f o[4][2];
#pragma unroll
  for (int mi = 0; mi < 4; ++mi) { o[mi][0] = (v8f){0,0,0,0,0,0,0,0};
                                   o[mi][1] = (v8f){0,0,0,0,0,0,0,0}; }
#pragma unroll
  for (int kt2 = 0; kt2 < 2; ++kt2) {
    v16h vt0 = tile_ld(vb, 1, C3, 0,  kt2 * 32, lane);  // row = channel
    v16h vt1 = tile_ld(vb, 1, C3, 16, kt2 * 32, lane);
#pragma unroll
    for (int mi = 0; mi < 4; ++mi) {
      v16h pa = tile_ld(P, 64, 1, mi * 16, kt2 * 32, lane);
      o[mi][0] = wmma_f16(pa, vt0, o[mi][0]);
      o[mi][1] = wmma_f16(pa, vt1, o[mi][1]);
    }
  }
  h16* ob = aout + (size_t)win * 64 * C + head * 32;
#pragma unroll
  for (int mi = 0; mi < 4; ++mi)
#pragma unroll
    for (int ci = 0; ci < 2; ++ci)
#pragma unroll
      for (int v = 0; v < 8; ++v) {
        int rq = mi * 16 + hi * 8 + v;
        if (rq < 49) ob[(size_t)rq * C + ci * 16 + ln] = (h16)o[mi][ci][v];
      }
  // zero padded rows so the proj GEMM sees finite inputs
  for (int rq = 49; rq < 64; ++rq)
    ob[(size_t)rq * C + lane] = (h16)0.f;
}

// scatter windowed proj result back to canonical layout (residual add)
__global__ __launch_bounds__(128)
void k_scatter_add(const h16* __restrict__ pout, float* __restrict__ X, int Hs,
                   int C, int shift, int nWs) {
  const int tok = blockIdx.x;                  // B*Hs*Hs
  const int bb = tok / (Hs * Hs);
  const int rm = tok % (Hs * Hs);
  const int hh = rm / Hs, ww = rm % Hs;
  const int rh = (hh - shift + Hs) % Hs;
  const int rw = (ww - shift + Hs) % Hs;
  const int wi = rh / 7, r = rh % 7, wj = rw / 7, cc = rw % 7;
  const size_t win = ((size_t)bb * nWs + wi) * nWs + wj;
  const int p = r * 7 + cc;
  const h16* src = pout + (win * 64 + p) * C;
  float* dst = X + (size_t)tok * C;
  for (int c = threadIdx.x; c < C; c += 128) dst[c] += (float)src[c];
}

__global__ void k_add_res(const h16* __restrict__ src, float* __restrict__ X,
                          long total) {
  long i = (long)blockIdx.x * blockDim.x + threadIdx.x;
  long stride = (long)gridDim.x * blockDim.x;
  for (; i < total; i += stride) X[i] += (float)src[i];
}

// 2x2 gather + LN over 4C -> f16 [T/4, 4C]
__global__ __launch_bounds__(128)
void k_merge_ln(const float* __restrict__ X, const float* __restrict__ g,
                const float* __restrict__ b, h16* __restrict__ out, int Hs,
                int C) {
  const int H2 = Hs / 2;
  const int tok = blockIdx.x;                  // B*H2*H2
  const int bb = tok / (H2 * H2);
  const int rm = tok % (H2 * H2);
  const int i = rm / H2, j = rm % H2;
  const int tid = threadIdx.x;
  const int C4 = 4 * C;
  __shared__ float s1[128], s2[128];
  float a = 0.f, q = 0.f;
  for (int f = tid; f < C4; f += 128) {
    int d = f / C, c = f % C;
    int di = d >> 1, dj = d & 1;
    float v = X[(((size_t)bb * Hs + 2 * i + di) * Hs + 2 * j + dj) * C + c];
    a += v; q += v * v;
  }
  s1[tid] = a; s2[tid] = q; __syncthreads();
  for (int st = 64; st > 0; st >>= 1) {
    if (tid < st) { s1[tid] += s1[tid + st]; s2[tid] += s2[tid + st]; }
    __syncthreads();
  }
  float mean = s1[0] / C4, var = s2[0] / C4 - mean * mean;
  float inv = rsqrtf(var + 1e-5f);
  h16* dst = out + (size_t)tok * C4;
  for (int f = tid; f < C4; f += 128) {
    int d = f / C, c = f % C;
    int di = d >> 1, dj = d & 1;
    float v = X[(((size_t)bb * Hs + 2 * i + di) * Hs + 2 * j + dj) * C + c];
    dst[f] = (h16)((v - mean) * inv * g[f] + b[f]);
  }
}

// bilinear 2x downsample (== 2x2 avg with half-pixel centers) + NHWC->NCHW
__global__ void k_pool_out(const float* __restrict__ feat, float* __restrict__ out,
                           int Hs, int C) {
  const int ts = Hs / 2;
  const size_t total = (size_t)8 * C * ts * ts;
  for (size_t i = (size_t)blockIdx.x * blockDim.x + threadIdx.x; i < total;
       i += (size_t)gridDim.x * blockDim.x) {
    int ow = (int)(i % ts);
    size_t t = i / ts;
    int oh = (int)(t % ts); t /= ts;
    int c = (int)(t % C);
    int b = (int)(t / C);
    float s = 0.f;
#pragma unroll
    for (int di = 0; di < 2; ++di)
#pragma unroll
      for (int dj = 0; dj < 2; ++dj)
        s += feat[(((size_t)b * Hs + 2 * oh + di) * Hs + 2 * ow + dj) * C + c];
    out[i] = 0.25f * s;
  }
}

// ---------------------------------------------------------------------------
// Host side
// ---------------------------------------------------------------------------
struct BW {
  const float *fc1_b, *fc1_w, *fc2_b, *fc2_w, *ln1_b, *ln1_g, *ln2_b, *ln2_g,
      *proj_b, *proj_w, *qkv_b, *qkv_w, *rel;
};

extern "C" void kernel_launch(void* const* d_in, const int* in_sizes, int n_in,
                              void* d_out, int out_size, void* d_ws, size_t ws_size,
                              hipStream_t stream) {
  (void)in_sizes; (void)n_in; (void)out_size; (void)ws_size;
  // ---- unpack inputs (jax tree-leaf order: dict keys sorted per level) ----
  int idx = 0;
  auto nf = [&]() -> const float* { return (const float*)d_in[idx++]; };
  const float* pe_b    = nf();
  const float* pe_ln_b = nf();
  const float* pe_ln_g = nf();
  const float* pe_w    = nf();
  BW blk[10]; int nb = 0;
  const float *ds_ln_b[2], *ds_ln_g[2], *ds_red[2];
  const int depths[3] = {2, 2, 6};
  for (int s = 0; s < 3; ++s) {
    for (int d = 0; d < depths[s]; ++d) {
      BW& w = blk[nb++];
      w.fc1_b = nf(); w.fc1_w = nf(); w.fc2_b = nf(); w.fc2_w = nf();
      w.ln1_b = nf(); w.ln1_g = nf(); w.ln2_b = nf(); w.ln2_g = nf();
      w.proj_b = nf(); w.proj_w = nf(); w.qkv_b = nf(); w.qkv_w = nf();
      w.rel = nf();
    }
    if (s < 2) { ds_ln_b[s] = nf(); ds_ln_g[s] = nf(); ds_red[s] = nf(); }
  }
  const float* xin = nf();

  // ---- workspace layout ----
  size_t cur = 0;
  auto alloc = [&](size_t sz) -> char* {
    char* p = (char*)d_ws + cur;
    cur += (sz + 255) & ~(size_t)255;
    return p;
  };
  float* X      = (float*)alloc((size_t)25088 * 96 * 4);
  float* feat0  = (float*)alloc((size_t)25088 * 96 * 4);
  float* feat1  = (float*)alloc((size_t)6272 * 192 * 4);
  float* feat2  = (float*)alloc((size_t)1568 * 384 * 4);
  h16*   XW     = (h16*)alloc((size_t)512 * 64 * 96 * 2);   // 6.29 MB (max)
  h16*   BIG    = (h16*)alloc((size_t)25088 * 384 * 2);     // QKV / MLP hidden
  h16*   AOUT   = (h16*)alloc((size_t)512 * 64 * 96 * 2);   // attn out / ln2
  h16*   POUT   = (h16*)alloc((size_t)512 * 64 * 96 * 2);   // proj / mlp out
  float* feats[3] = {feat0, feat1, feat2};

  // ---- forward ----
  k_patch_embed<<<25088, 128, 0, stream>>>(xin, pe_w, pe_b, pe_ln_g, pe_ln_b, X);

  const int HEADS[3] = {3, 6, 12};
  int Hs = 56, nb2 = 0;
  for (int s = 0; s < 3; ++s) {
    const int C = 96 << s, nH = HEADS[s];
    const int nWs = Hs / 7;
    const int Bn = 8 * nWs * nWs;
    const int Mw = Bn * 64;
    const int T = 8 * Hs * Hs;
    for (int d = 0; d < depths[s]; ++d) {
      BW& w = blk[nb2++];
      const int shift = (d & 1) ? 3 : 0;
      // --- attention branch ---
      k_ln_part<<<Mw, 128, 0, stream>>>(X, w.ln1_g, w.ln1_b, XW, Hs, C, shift, nWs);
      k_gemm<<<dim3((Mw + BM - 1) / BM, (3 * C + BN - 1) / BN), 256, 0, stream>>>(
          XW, w.qkv_w, w.qkv_b, BIG, nullptr, Mw, 3 * C, C, 0);
      k_attn<<<Bn, nH * 32, (size_t)nH * 64 * 64 * 2, stream>>>(
          BIG, AOUT, w.rel, C, nH, shift, nWs);
      k_gemm<<<dim3((Mw + BM - 1) / BM, (C + BN - 1) / BN), 256, 0, stream>>>(
          AOUT, w.proj_w, w.proj_b, POUT, nullptr, Mw, C, C, 0);
      k_scatter_add<<<T, 128, 0, stream>>>(POUT, X, Hs, C, shift, nWs);
      // --- MLP branch ---
      k_ln_tok<<<T, 128, 0, stream>>>(X, w.ln2_g, w.ln2_b, AOUT, C);
      k_gemm<<<dim3((T + BM - 1) / BM, (4 * C + BN - 1) / BN), 256, 0, stream>>>(
          AOUT, w.fc1_w, w.fc1_b, BIG, nullptr, T, 4 * C, C, /*GELU*/ 1);
      k_gemm<<<dim3((T + BM - 1) / BM, (C + BN - 1) / BN), 256, 0, stream>>>(
          BIG, w.fc2_w, w.fc2_b, POUT, nullptr, T, C, 4 * C, 0);
      long tot = (long)T * C;
      k_add_res<<<(int)((tot + 255) / 256), 256, 0, stream>>>(POUT, X, tot);
    }
    (void)hipMemcpyAsync(feats[s], X, (size_t)T * C * 4,
                         hipMemcpyDeviceToDevice, stream);
    if (s < 2) {
      const int T4 = T / 4;
      k_merge_ln<<<T4, 128, 0, stream>>>(X, ds_ln_g[s], ds_ln_b[s], XW, Hs, C);
      k_gemm<<<dim3((T4 + BM - 1) / BM, (2 * C + BN - 1) / BN), 256, 0, stream>>>(
          XW, ds_red[s], nullptr, nullptr, X, T4, 2 * C, 4 * C, /*f32 out*/ 2);
      Hs /= 2;
    }
  }
  // ---- outputs: bilinear 2x down (== avg-pool 2x2) + NHWC->NCHW ----
  float* out = (float*)d_out;
  k_pool_out<<<2352, 256, 0, stream>>>(feat0, out, 56, 96);
  k_pool_out<<<1176, 256, 0, stream>>>(feat1, out + 602112, 28, 192);
  k_pool_out<<<588, 256, 0, stream>>>(feat2, out + 602112 + 301056, 14, 384);
}
